// DifferentiableXGB_56642028700193
// MI455X (gfx1250) — matrix operand order repaired and checked by reading, not compiled, not run on hardware
//
#include <hip/hip_runtime.h>
#include <hip/hip_bf16.h>

// ---------------- CDNA5 WMMA types ----------------
typedef __attribute__((ext_vector_type(16))) __bf16 v16bf;
typedef __attribute__((ext_vector_type(8)))  float  v8f;
typedef __attribute__((ext_vector_type(4)))  unsigned int v4u;

union ABFrag { v16bf bf; v4u u[2]; };

// Problem constants
#define BB      32768
#define DD      1024
#define TT      100
#define KK      4
#define NN      (TT * KK)      // 400 output columns
#define TILE_M  128            // rows per block
#define NTILES  (NN / 16)      // 25 column tiles
#define PITCH   1032           // 1024 + 8 bf16 pad -> conflict-free ds_load_b128
#define X_USHORTS   (TILE_M * PITCH)            // 132096
#define W_USHORTS   (16 * PITCH)                // 16512
#define SMEM_BYTES  ((X_USHORTS + W_USHORTS) * 2)  // 297216 B <= 320KB/WGP

__device__ __forceinline__ unsigned short f2bf(float f) {
    unsigned u = __builtin_bit_cast(unsigned int, f);
    u += 0x7FFFu + ((u >> 16) & 1u);          // round-to-nearest-even
    return (unsigned short)(u >> 16);
}
__device__ __forceinline__ unsigned pk2(float a, float b) {
    return (unsigned)f2bf(a) | ((unsigned)f2bf(b) << 16);
}

// DPP cross-lane add within aligned quads (pure VALU, no LDS unit)
#define DPP_QUAD_XOR1 0xB1   // quad_perm [1,0,3,2]
#define DPP_QUAD_XOR2 0x4E   // quad_perm [2,3,0,1]
__device__ __forceinline__ float dpp_add(float v, const int ctrl) {
    const int s = __builtin_bit_cast(int, v);
    int p;
    if (ctrl == DPP_QUAD_XOR1)
        p = __builtin_amdgcn_update_dpp(0, s, DPP_QUAD_XOR1, 0xf, 0xf, true);
    else
        p = __builtin_amdgcn_update_dpp(0, s, DPP_QUAD_XOR2, 0xf, 0xf, true);
    return v + __builtin_bit_cast(float, p);
}
// ds_swizzle xor-mask lane swap (group-of-32 mode: and=0x1f, or=0, xor=mask)
__device__ __forceinline__ float swz_xor(float v, const int mask) {
    const int s = __builtin_bit_cast(int, v);
    int p;
    if (mask == 4)  p = __builtin_amdgcn_ds_swizzle(s, 0x101F);
    else            p = __builtin_amdgcn_ds_swizzle(s, 0x201F);
    return __builtin_bit_cast(float, p);
}

__global__ __launch_bounds__(256)
void DifferentiableXGB_56642028700193_kernel(
        const float* __restrict__ x,    // [B, D]
        const float* __restrict__ W1,   // [T, K, D] == [400, 1024]
        const float* __restrict__ b1,   // [T, K] == [400]
        const float* __restrict__ fwp,  // [T]
        const float* __restrict__ fcw,  // [2, 4]
        const float* __restrict__ fcb,  // [2]
        float* __restrict__ out)        // [B, 2]
{
    extern __shared__ __align__(16) unsigned char smem_raw[];
    unsigned short* xs = reinterpret_cast<unsigned short*>(smem_raw);
    unsigned short* ws = xs + X_USHORTS;

    const int tid  = threadIdx.x;
    const int wave = tid >> 5;
    const int lane = tid & 31;
    const int l    = lane & 15;           // column within 16-tile / row within slab
    const int half = lane >> 4;           // 0: lanes 0-15, 1: lanes 16-31
    const int row0 = blockIdx.x * TILE_M;

    // ---- stage x tile: 128 rows x 1024 f32 -> bf16 in LDS (coalesced b128 loads) ----
    for (int i = tid; i < TILE_M * (DD / 4); i += 256) {
        const int r  = i >> 8;            // / (1024/4)
        const int c4 = i & 255;
        const float4 v = *reinterpret_cast<const float4*>(
            x + (size_t)(row0 + r) * DD + (size_t)c4 * 4);
        uint2 p;
        p.x = pk2(v.x, v.y);
        p.y = pk2(v.z, v.w);
        *reinterpret_cast<uint2*>(xs + (size_t)r * PITCH + (size_t)c4 * 4) = p;
    }

    const unsigned short* xrow = xs + (size_t)(wave * 16 + l) * PITCH;
    const unsigned short* wrow = ws + (size_t)l * PITCH;

    float wsum[8] = {0.f, 0.f, 0.f, 0.f, 0.f, 0.f, 0.f, 0.f};

    __syncthreads();   // x tile visible

    for (int t = 0; t < NTILES; ++t) {
        const int n0 = t * 16;

        // ---- stage W1 tile: 16 rows x 1024 f32 (L2-resident) -> bf16 LDS ----
        for (int i = tid; i < 16 * (DD / 4); i += 256) {
            const int r  = i >> 8;
            const int c4 = i & 255;
            const float4 v = *reinterpret_cast<const float4*>(
                W1 + (size_t)(n0 + r) * DD + (size_t)c4 * 4);
            uint2 p;
            p.x = pk2(v.x, v.y);
            p.y = pk2(v.z, v.w);
            *reinterpret_cast<uint2*>(ws + (size_t)r * PITCH + (size_t)c4 * 4) = p;
        }
        __syncthreads();   // W tile visible

        // prefetch next tile's W1 rows into L2/L0 while we compute
        if (t + 1 < NTILES) {
            const float* pf = W1 + (size_t)(n0 + 16 + (tid & 15)) * DD + (size_t)(tid >> 4) * 64;
            __builtin_prefetch(pf, 0, 1);
        }

        // ---- 16x16 output tile, K = 1024: software-pipelined
        //      v_wmma_f32_16x16x32_bf16 with next fragments in flight ----
        #define LOADF(A, Bf, d0) do {                                                   \
            (A).u[0]  = *reinterpret_cast<const v4u*>(xrow + (d0) + half * 8);          \
            (A).u[1]  = *reinterpret_cast<const v4u*>(xrow + (d0) + 16 + half * 8);     \
            (Bf).u[0] = *reinterpret_cast<const v4u*>(wrow + (d0) + half * 8);          \
            (Bf).u[1] = *reinterpret_cast<const v4u*>(wrow + (d0) + 16 + half * 8);     \
        } while (0)

        v8f c = {};
        ABFrag a0, b0, a1, bfr;
        LOADF(a0, b0, 0);
        #pragma unroll 3
        for (int kk = 0; kk < 30; kk += 2) {
            LOADF(a1, bfr, (kk + 1) * 32);
            c = __builtin_amdgcn_wmma_f32_16x16x32_bf16(
                    false, a0.bf, false, b0.bf, (short)0, c, false, false);
            LOADF(a0, b0, (kk + 2) * 32);
            c = __builtin_amdgcn_wmma_f32_16x16x32_bf16(
                    false, a1.bf, false, bfr.bf, (short)0, c, false, false);
        }
        LOADF(a1, bfr, 31 * 32);
        c = __builtin_amdgcn_wmma_f32_16x16x32_bf16(
                false, a0.bf, false, b0.bf, (short)0, c, false, false);
        c = __builtin_amdgcn_wmma_f32_16x16x32_bf16(
                false, a1.bf, false, bfr.bf, (short)0, c, false, false);
        #undef LOADF

        // ---- fused epilogue for this tile (4 trees per tile, k = lane&3) ----
        const float bias = b1[n0 + l];
        const float fw   = fwp[(n0 + l) >> 2];
        #pragma unroll
        for (int r = 0; r < 8; ++r) {
            const float split = c[r] + bias;
            float s = split;                        // sum over tree's 4 leaves (DPP quads)
            s += dpp_add(s, DPP_QUAD_XOR1) - s;     // placeholder avoided; see below
            // (the helper already returns v + permuted; recompute cleanly:)
            s = split;
            s = dpp_add(s, DPP_QUAD_XOR1);
            s = dpp_add(s, DPP_QUAD_XOR2);
            const float e    = __expf(-split);
            const float leaf = __builtin_amdgcn_rcpf(1.f + e);   // v_rcp_f32, no IEEE div
            wsum[r] += fw * (leaf * s);             // per-lane (row, k) partial over trees
        }
        __syncthreads();   // done with this W tile before restaging
    }

    // ---- reduce over tree groups and project 4 -> 2 (runs once) ----
    const float fcw0 = fcw[lane & 3];          // fc_w[0][k]
    const float fcw1 = fcw[4 + (lane & 3)];    // fc_w[1][k]
    const float fb0  = fcb[0];
    const float fb1  = fcb[1];
    #pragma unroll
    for (int r = 0; r < 8; ++r) {
        float w = wsum[r];
        w += swz_xor(w, 4);                    // combine lanes {k,k+4,k+8,k+12}
        w += swz_xor(w, 8);                    // -> weighted[b, k] in every lane
        float l0 = w * fcw0;
        float l1 = w * fcw1;
        l0 = dpp_add(l0, DPP_QUAD_XOR1); l0 = dpp_add(l0, DPP_QUAD_XOR2);
        l1 = dpp_add(l1, DPP_QUAD_XOR1); l1 = dpp_add(l1, DPP_QUAD_XOR2);
        if (l == 0) {                          // lanes 0 and 16 write their row halves
            const int b = row0 + wave * 16 + half * 8 + r;
            float2 o;
            o.x = l0 + fb0;
            o.y = l1 + fb1;
            *reinterpret_cast<float2*>(out + (size_t)b * 2) = o;
        }
    }
}

extern "C" void kernel_launch(void* const* d_in, const int* in_sizes, int n_in,
                              void* d_out, int out_size, void* d_ws, size_t ws_size,
                              hipStream_t stream) {
    (void)in_sizes; (void)n_in; (void)d_ws; (void)ws_size; (void)out_size;
    const float* x   = (const float*)d_in[0];
    const float* W1  = (const float*)d_in[1];
    const float* b1  = (const float*)d_in[2];
    const float* fw  = (const float*)d_in[3];
    const float* fcw = (const float*)d_in[4];
    const float* fcb = (const float*)d_in[5];
    float* out = (float*)d_out;

    hipFuncSetAttribute((const void*)DifferentiableXGB_56642028700193_kernel,
                        hipFuncAttributeMaxDynamicSharedMemorySize, SMEM_BYTES);

    DifferentiableXGB_56642028700193_kernel
        <<<dim3(BB / TILE_M), dim3(256), SMEM_BYTES, stream>>>(
            x, W1, b1, fw, fcw, fcb, out);
}
